// ProxyNCA_prob_match_40664750359183
// MI455X (gfx1250) — matrix operand order repaired and checked by reading, not compiled, not run on hardware
//
#include <hip/hip_runtime.h>
#include <hip/hip_bf16.h>
#include <math.h>

// ---------------------------------------------------------------------------
// ProxyNCA-style loss, MI455X (gfx1250) wave32: WMMA bf16 + TDM double-buffer.
// B=512, C=100000, D=128.
// ---------------------------------------------------------------------------

#define B_ROWS   512
#define C_PROX   100000
#define D_EMB    128
#define NCHUNK   50
#define COLTILES 6250            // C_PROX / 16
#define TILES_PER_CHUNK 125      // COLTILES / NCHUNK
#define NEG_BIG  (-3.0e38f)

typedef __attribute__((ext_vector_type(16))) __bf16 v16bf;
typedef __attribute__((ext_vector_type(8)))  __bf16 v8bf;
typedef __attribute__((ext_vector_type(4)))  __bf16 v4bf;
typedef __attribute__((ext_vector_type(8)))  float  v8f;
typedef __attribute__((ext_vector_type(4)))  unsigned int u32x4;
typedef __attribute__((ext_vector_type(8)))  int          i32x8;
typedef __attribute__((ext_vector_type(4)))  int          i32x4;

// ---- WMMA fragment loaders (contiguous K order, same for A and B, wave32) ----------
// A (16x32 bf16, MxK): lane lr=lane&15 holds row M=lr.
//   half==0: elems 0..15 = K kk+0..15
//   half==1: elems 0..15 = K kk+16..31
__device__ __forceinline__ v16bf load_a_frag(const __bf16* src, int row, int kk, int half) {
    const __bf16* p0 = src + row * D_EMB + kk + half * 16;  // 16B aligned
    const __bf16* p1 = p0 + 8;
    v8bf lo = *(const v8bf*)p0;
    v8bf hi = *(const v8bf*)p1;
    v16bf a;
#pragma unroll
    for (int i = 0; i < 8; ++i) { a[i] = lo[i]; a[8 + i] = hi[i]; }
    return a;
}

// B (32x16 bf16, KxN): lane lr holds column N=lr.
//   lanes 0-15: elems 0..15 = K kk+0..15 ; lanes 16-31: K kk+16..31
__device__ __forceinline__ v16bf load_b_frag(const __bf16* src, int col, int kk, int half) {
    return *(const v16bf*)(src + col * D_EMB + kk + half * 16);   // 32B aligned
}

// ---- Tensor Data Mover: stage one 16x128 bf16 tile (4KB) into LDS ----------
// D# per CDNA5 ISA: 2D tensor, data_size=2B,
// tile_dim0=128 (K), tile_dim1=16 (rows), tensor_dim0_stride=128, type=2.
__device__ __forceinline__ void tdm_load_tile(unsigned lds_off, const __bf16* gptr) {
    unsigned long long ga = (unsigned long long)(uintptr_t)gptr;
    u32x4 g0;
    g0[0] = 1u;                                               // count=1 (valid), user mode
    g0[1] = lds_off;                                          // lds_addr (bytes)
    g0[2] = (unsigned)(ga & 0xFFFFFFFFu);                     // global_addr[31:0]
    g0[3] = (unsigned)((ga >> 32) & 0x01FFFFFFu) | (2u << 30);// global_addr[56:32] | type=2
    i32x8 g1;
    g1[0] = (1 << 16);        // workgroup_mask=0, data_size=1 (2 bytes), no pad/iterate
    g1[1] = (D_EMB << 16);    // tensor_dim0[15:0]=128 in bits[63:48]
    g1[2] = (16 << 16);       // tensor_dim0[31:16]=0 ; tensor_dim1[15:0]=16
    g1[3] = (D_EMB << 16);    // tensor_dim1[31:16]=0 ; tile_dim0=128
    g1[4] = 16;               // tile_dim1=16 ; tile_dim2=0
    g1[5] = D_EMB;            // tensor_dim0_stride[31:0]=128
    g1[6] = 0;                // stride0 hi ; tensor_dim1_stride lo
    g1[7] = 0;                // tensor_dim1_stride hi
    i32x4 z4 = {0, 0, 0, 0};
#if __has_include(<hip/amd_detail/amd_gfx1250_TDM.h>)
    i32x8 z8 = {0, 0, 0, 0, 0, 0, 0, 0};
    __builtin_amdgcn_tensor_load_to_lds(g0, g1, z4, z4, z8, 0);   // clang-23 6-arg form
#else
    __builtin_amdgcn_tensor_load_to_lds(g0, g1, z4, z4, 0);       // ROCm 7.2 5-arg form
#endif
}

// ---------------------------------------------------------------------------
// Kernel 1: Xn = l2norm(X); Pb[b] = l2norm(proxies[T[b]]); diag[b]=<Pb,Xn>.
// Also writes bf16 copies Xnb, Pbb for the WMMA attention kernel.
// ---------------------------------------------------------------------------
__global__ __launch_bounds__(128) void prep_kernel(const float* __restrict__ X,
                                                   const int*   __restrict__ T,
                                                   const float* __restrict__ proxies,
                                                   float* __restrict__ Xn,
                                                   float* __restrict__ Pb,
                                                   float* __restrict__ diag,
                                                   __bf16* __restrict__ Xnb,
                                                   __bf16* __restrict__ Pbb) {
    const int b = blockIdx.x, d = threadIdx.x;
    __shared__ float red[128];

    float x = X[b * D_EMB + d];
    red[d] = x * x;
    __syncthreads();
#pragma unroll
    for (int off = 64; off > 0; off >>= 1) { if (d < off) red[d] += red[d + off]; __syncthreads(); }
    float xn = x / fmaxf(sqrtf(red[0]), 1e-12f);
    Xn[b * D_EMB + d] = xn;
    Xnb[b * D_EMB + d] = (__bf16)xn;
    __syncthreads();

    const int t = T[b];
    float p = proxies[(long)t * D_EMB + d];
    red[d] = p * p;
    __syncthreads();
#pragma unroll
    for (int off = 64; off > 0; off >>= 1) { if (d < off) red[d] += red[d + off]; __syncthreads(); }
    float pb = p / fmaxf(sqrtf(red[0]), 1e-12f);
    Pb[b * D_EMB + d] = pb;
    Pbb[b * D_EMB + d] = (__bf16)pb;
    __syncthreads();

    red[d] = pb * xn;
    __syncthreads();
#pragma unroll
    for (int off = 64; off > 0; off >>= 1) { if (d < off) red[d] += red[d + off]; __syncthreads(); }
    if (d == 0) diag[b] = red[0];
}

// ---------------------------------------------------------------------------
// Kernel 2: Psb[c,:] = bf16( proxies[c,:] * 3 / max(||proxies[c]||,1e-12) ).
// One wave per row; the single mandatory HBM sweep of the 51.2MB proxy array.
// ---------------------------------------------------------------------------
__global__ __launch_bounds__(256) void pscale_kernel(const float* __restrict__ proxies,
                                                     __bf16* __restrict__ Psb) {
    const int wave = threadIdx.x >> 5, lane = threadIdx.x & 31;
    const int row = blockIdx.x * 8 + wave;
    const float* p = proxies + (long)row * D_EMB + lane * 4;
    float x0 = p[0], x1 = p[1], x2 = p[2], x3 = p[3];
    float s = x0 * x0 + x1 * x1 + x2 * x2 + x3 * x3;
#pragma unroll
    for (int off = 16; off >= 1; off >>= 1) s += __shfl_xor(s, off, 32);
    const float scl = 3.0f / fmaxf(sqrtf(s), 1e-12f);
    v4bf o;
    o[0] = (__bf16)(x0 * scl); o[1] = (__bf16)(x1 * scl);
    o[2] = (__bf16)(x2 * scl); o[3] = (__bf16)(x3 * scl);
    *(v4bf*)(Psb + (long)row * D_EMB + lane * 4) = o;
}

// ---------------------------------------------------------------------------
// Kernel 3: att[b,k] = relu(0.5*(<Pb_b,Xn_k> + diag[k])) + relu(<Xn_b,Xn_k>)
// 1024 16x16 tiles; one wave per tile; both GEMMs share the B fragment.
// ---------------------------------------------------------------------------
__global__ __launch_bounds__(256) void att_kernel(const __bf16* __restrict__ Xnb,
                                                  const __bf16* __restrict__ Pbb,
                                                  const float* __restrict__ diag,
                                                  float* __restrict__ att) {
    const int wave = threadIdx.x >> 5, lane = threadIdx.x & 31;
    const int half = lane >> 4, lr = lane & 15;
    const int tile = blockIdx.x * 8 + wave;       // 0..1023
    const int b0 = (tile >> 5) * 16;              // row tile
    const int k0 = (tile & 31) * 16;              // col tile

    v8f acc1 = {}; v8f acc2 = {};
#pragma unroll
    for (int kk = 0; kk < D_EMB; kk += 32) {
        v16bf aPb = load_a_frag(Pbb, b0 + lr, kk, half);
        v16bf aXn = load_a_frag(Xnb, b0 + lr, kk, half);
        v16bf bXn = load_b_frag(Xnb, k0 + lr, kk, half);
        acc1 = __builtin_amdgcn_wmma_f32_16x16x32_bf16(false, aPb, false, bXn, (short)0, acc1, false, false);
        acc2 = __builtin_amdgcn_wmma_f32_16x16x32_bf16(false, aXn, false, bXn, (short)0, acc2, false, false);
    }
    const float dg = diag[k0 + lr];
#pragma unroll
    for (int v = 0; v < 8; ++v) {
        const int row = b0 + v + 8 * half;
        float a2b = fmaxf(0.5f * (acc1[v] + dg), 0.0f);
        float a2x = fmaxf(acc2[v], 0.0f);
        att[row * B_ROWS + k0 + lr] = a2b + a2x;
    }
}

// ---------------------------------------------------------------------------
// Kernel 4: inv[k] = 1 / (sum_j att[k,j] + 1e-5)   (torch column-divide quirk)
// ---------------------------------------------------------------------------
__global__ __launch_bounds__(128) void rowsum_kernel(const float* __restrict__ att,
                                                     float* __restrict__ inv) {
    const int k = blockIdx.x, t = threadIdx.x;
    __shared__ float red[128];
    float s = 0.0f;
    for (int j = t; j < B_ROWS; j += 128) s += att[k * B_ROWS + j];
    red[t] = s;
    __syncthreads();
#pragma unroll
    for (int off = 64; off > 0; off >>= 1) { if (t < off) red[t] += red[t + off]; __syncthreads(); }
    if (t == 0) inv[k] = 1.0f / (red[0] + 1e-5f);
}

// ---------------------------------------------------------------------------
// Kernel 5: aw = (att * inv) @ Xn ; X2 = Xn + aw ; Xs = 3*l2norm(X2) -> Xsb;
//           dmt[b] = max(18 - 6*<Xs_b, Pb_b>, 0)    (target-class distance)
// ---------------------------------------------------------------------------
__global__ __launch_bounds__(128) void aw_kernel(const float* __restrict__ att,
                                                 const float* __restrict__ inv,
                                                 const float* __restrict__ Xn,
                                                 const float* __restrict__ Pb,
                                                 __bf16* __restrict__ Xsb,
                                                 float* __restrict__ dmt) {
    const int b = blockIdx.x, d = threadIdx.x;
    __shared__ float s_arow[B_ROWS];
    __shared__ float red[128];

    for (int k = d; k < B_ROWS; k += 128) s_arow[k] = att[b * B_ROWS + k] * inv[k];
    __syncthreads();

    float acc = 0.0f;
    for (int k = 0; k < B_ROWS; ++k) acc = fmaf(s_arow[k], Xn[k * D_EMB + d], acc);
    const float x2 = Xn[b * D_EMB + d] + acc;

    red[d] = x2 * x2;
    __syncthreads();
#pragma unroll
    for (int off = 64; off > 0; off >>= 1) { if (d < off) red[d] += red[d + off]; __syncthreads(); }
    const float xs = 3.0f * x2 / fmaxf(sqrtf(red[0]), 1e-12f);
    Xsb[b * D_EMB + d] = (__bf16)xs;
    __syncthreads();

    red[d] = xs * Pb[b * D_EMB + d];
    __syncthreads();
#pragma unroll
    for (int off = 64; off > 0; off >>= 1) { if (d < off) red[d] += red[d + off]; __syncthreads(); }
    if (d == 0) dmt[b] = fmaxf(18.0f - 6.0f * red[0], 0.0f);
}

// ---------------------------------------------------------------------------
// Kernel 6: fused GEMM + online softmax, TDM double-buffered.
//   Dm[b,c] = max(18 - 2*<Xs_b, Ps_c>, 0); per-row (m,s) of exp(-Dm) per chunk.
//   Each wave streams its 16x128 bf16 proxy tiles Global->LDS via the Tensor
//   Data Mover (TENSORcnt), double-buffered in a private 2x4KB LDS slot.
// ---------------------------------------------------------------------------
__global__ __launch_bounds__(256) void bigloss_kernel(const __bf16* __restrict__ Xsb,
                                                      const __bf16* __restrict__ Psb,
                                                      float* __restrict__ pm,
                                                      float* __restrict__ ps) {
    __shared__ __bf16 tiles[8][2][16 * D_EMB];   // 8 waves x 2 slots x 4KB = 64KB
    __shared__ float lm[8][16];
    __shared__ float ls[8][16];

    const int chunk = blockIdx.x;          // 0..NCHUNK-1
    const int rowTile = blockIdx.y;        // 0..31
    const int wave = threadIdx.x >> 5, lane = threadIdx.x & 31;
    const int half = lane >> 4, lr = lane & 15;
    const int b0 = rowTile * 16;

    // Hoist A fragments (Xs rows b0..b0+15, all 128 K) out of the column loop.
    v16bf afr[4];
#pragma unroll
    for (int kk = 0; kk < 4; ++kk) afr[kk] = load_a_frag(Xsb, b0 + lr, kk * 32, half);

    float m[8], s[8];
#pragma unroll
    for (int v = 0; v < 8; ++v) { m[v] = NEG_BIG; s[v] = 0.0f; }

    const unsigned lds0 = (unsigned)(uintptr_t)&tiles[wave][0][0];
    const unsigned lds1 = (unsigned)(uintptr_t)&tiles[wave][1][0];
    const int tile0 = chunk * TILES_PER_CHUNK;

    // Prime the pipeline: wave's first tile into slot 0.
    tdm_load_tile(lds0, Psb + (long)(tile0 + wave) * 16 * D_EMB);

    int cur = 0;
    for (int t = wave; t < TILES_PER_CHUNK; t += 8) {   // wave-uniform loop
        const int nt = t + 8;
        if (nt < TILES_PER_CHUNK) {
            tdm_load_tile(cur ? lds0 : lds1, Psb + (long)(tile0 + nt) * 16 * D_EMB);
            __builtin_amdgcn_s_wait_tensorcnt(1);       // current tile landed
        } else {
            __builtin_amdgcn_s_wait_tensorcnt(0);
        }

        const __bf16* tile = &tiles[wave][cur][0];
        v8f acc = {};
#pragma unroll
        for (int kk = 0; kk < 4; ++kk) {
            v16bf bfrag = *(const v16bf*)(tile + lr * D_EMB + kk * 32 + half * 16);
            acc = __builtin_amdgcn_wmma_f32_16x16x32_bf16(false, afr[kk], false, bfrag,
                                                          (short)0, acc, false, false);
        }

        // Distance + branchless online-softmax update (per lane: 8 rows, 1 col).
#pragma unroll
        for (int v = 0; v < 8; ++v) {
            float dval = fmaxf(18.0f - 2.0f * acc[v], 0.0f);
            float x = -dval;
            float M = fmaxf(m[v], x);
            s[v] = s[v] * __expf(m[v] - M) + __expf(x - M);
            m[v] = M;
        }
        cur ^= 1;
    }

    // Reduce (m,s) across the 16 lanes (same rows) of each half-wave.
#pragma unroll
    for (int off = 8; off >= 1; off >>= 1) {
#pragma unroll
        for (int v = 0; v < 8; ++v) {
            float mo = __shfl_xor(m[v], off, 32);
            float so = __shfl_xor(s[v], off, 32);
            float M = fmaxf(m[v], mo);
            s[v] = s[v] * __expf(m[v] - M) + so * __expf(mo - M);
            m[v] = M;
        }
    }

    if (lr == 0) {
#pragma unroll
        for (int v = 0; v < 8; ++v) { lm[wave][half * 8 + v] = m[v]; ls[wave][half * 8 + v] = s[v]; }
    }
    __syncthreads();

    if (threadIdx.x < 16) {
        const int r = threadIdx.x;
        float M = NEG_BIG, S = 0.0f;
#pragma unroll
        for (int w = 0; w < 8; ++w) {
            float mo = lm[w][r], so = ls[w][r];
            float Mn = fmaxf(M, mo);
            S = S * __expf(M - Mn) + so * __expf(mo - Mn);
            M = Mn;
        }
        pm[chunk * B_ROWS + b0 + r] = M;
        ps[chunk * B_ROWS + b0 + r] = S;
    }
}

// ---------------------------------------------------------------------------
// Kernel 7: combine chunk partials, loss[b] = dmt[b] + (M + log S), mean.
// ---------------------------------------------------------------------------
__global__ __launch_bounds__(512) void finalize_kernel(const float* __restrict__ pm,
                                                       const float* __restrict__ ps,
                                                       const float* __restrict__ dmt,
                                                       float* __restrict__ out) {
    const int b = threadIdx.x;   // 512 threads, 1 block
    float M = NEG_BIG, S = 0.0f;
    for (int c = 0; c < NCHUNK; ++c) {
        float mo = pm[c * B_ROWS + b], so = ps[c * B_ROWS + b];
        float Mn = fmaxf(M, mo);
        S = S * __expf(M - Mn) + so * __expf(mo - Mn);
        M = Mn;
    }
    float loss = dmt[b] + M + __logf(S);

    __shared__ float red[512];
    red[b] = loss;
    __syncthreads();
#pragma unroll
    for (int off = 256; off > 0; off >>= 1) { if (b < off) red[b] += red[b + off]; __syncthreads(); }
    if (b == 0) out[0] = red[0] / (float)B_ROWS;
}

// ---------------------------------------------------------------------------
// Launcher
// ---------------------------------------------------------------------------
extern "C" void kernel_launch(void* const* d_in, const int* in_sizes, int n_in,
                              void* d_out, int out_size, void* d_ws, size_t ws_size,
                              hipStream_t stream) {
    const float* X       = (const float*)d_in[0];   // (512,128)
    // d_in[1]: indices (unused)
    const int*   T       = (const int*)d_in[2];     // (512,)
    const float* proxies = (const float*)d_in[3];   // (100000,128)
    float* out = (float*)d_out;

    // Workspace layout: f32 region then bf16 region.
    float* W     = (float*)d_ws;
    float* Xn    = W;                 // 65536
    float* Pb    = Xn + 65536;        // 65536
    float* att   = Pb + 65536;        // 262144  (512x512)
    float* diag  = att + 262144;      // 512
    float* inv   = diag + 512;        // 512
    float* dmt   = inv + 512;         // 512
    float* pm    = dmt + 512;         // 25600  (NCHUNK*512)
    float* ps    = pm + 25600;        // 25600
    __bf16* Xnb  = (__bf16*)(ps + 25600);          // 65536 bf16
    __bf16* Pbb  = Xnb + 65536;                    // 65536 bf16
    __bf16* Xsb  = Pbb + 65536;                    // 65536 bf16
    __bf16* Psb  = Xsb + 65536;                    // 12,800,000 bf16 (25.6MB)
    (void)in_sizes; (void)n_in; (void)out_size; (void)ws_size;

    prep_kernel<<<B_ROWS, 128, 0, stream>>>(X, T, proxies, Xn, Pb, diag, Xnb, Pbb);
    pscale_kernel<<<C_PROX / 8, 256, 0, stream>>>(proxies, Psb);
    att_kernel<<<128, 256, 0, stream>>>(Xnb, Pbb, diag, att);
    rowsum_kernel<<<B_ROWS, 128, 0, stream>>>(att, inv);
    aw_kernel<<<B_ROWS, 128, 0, stream>>>(att, inv, Xn, Pb, Xsb, dmt);
    dim3 grid(NCHUNK, B_ROWS / 16);
    bigloss_kernel<<<grid, 256, 0, stream>>>(Xsb, Psb, pm, ps);
    finalize_kernel<<<1, 512, 0, stream>>>(pm, ps, dmt, out);
}
